// DeformableAtt_36558761623689
// MI455X (gfx1250) — compile-verified, matrix operand order checked by reading, not run on hardware
//
#include <hip/hip_runtime.h>

typedef __attribute__((ext_vector_type(16))) _Float16 v16h;
typedef __attribute__((ext_vector_type(8)))  _Float16 h8;
typedef __attribute__((ext_vector_type(4)))  _Float16 h4;
typedef __attribute__((ext_vector_type(8)))  float    v8f;
typedef __attribute__((ext_vector_type(4)))  float    f4;

#define ZV8 {0.f,0.f,0.f,0.f,0.f,0.f,0.f,0.f}

// ---------------------------------------------------------------------------
// WMMA fragment helpers (CDNA5 16x16x32 f16 -> f32)
// A (16x32) layout per ISA 7.12.2: lane<16: row=lane, K={0..7,16..23};
// lane>=16: row=lane-16, K={8..15,24..31}. Two contiguous 16B runs per lane.
// ---------------------------------------------------------------------------
__device__ __forceinline__ v16h frag_from_runs(h8 lo, h8 hi) {
  return __builtin_shufflevector(lo, hi, 0, 1, 2, 3, 4, 5, 6, 7,
                                 8, 9, 10, 11, 12, 13, 14, 15);
}

__device__ __forceinline__ v16h load_frag_a(const _Float16* __restrict__ p,
                                            int ld, int lane) {
  const int r  = lane & 15;
  const int kb = (lane & 16) ? 8 : 0;
  const _Float16* q = p + (size_t)r * ld + kb;
  h8 lo = *(const h8*)(q);
  h8 hi = *(const h8*)(q + 16);
  return frag_from_runs(lo, hi);
}

// packed B tile: [tile][lane][16] halves, one 32B contiguous load per lane
__device__ __forceinline__ v16h load_frag_bp(const _Float16* __restrict__ P,
                                             size_t tile, int lane) {
  return *(const v16h*)(P + tile * 512 + lane * 16);
}

__device__ __forceinline__ v8f wmma_f32(v16h a, v16h b, v8f c) {
  return __builtin_amdgcn_wmma_f32_16x16x32_f16(false, a, false, b,
                                                (short)0, c, false, false);
}

// CDNA5 async global->LDS copy, 16 bytes per lane (ASYNCcnt path).
// ldsoff = LDS byte address (generic shared ptr truncated per ISA addr[31:0]).
__device__ __forceinline__ void async_copy_b128(unsigned ldsoff,
                                                const void* gaddr) {
  asm volatile("global_load_async_to_lds_b128 %0, %1, off"
               :: "v"(ldsoff), "v"(gaddr) : "memory");
}
__device__ __forceinline__ void async_wait0() {
  asm volatile("s_wait_asynccnt 0x0" ::: "memory");
}

// ---------------------------------------------------------------------------
// K0: pack weight W (N rows x K cols f32, row-major: W[n*K+k]) into
// fragment-ready f16 tiles: P[((k0/32)*(N/16)+n0/16)*512 + lane*16 + i]
// ---------------------------------------------------------------------------
__global__ void dat_pack_b(const float* __restrict__ W, _Float16* __restrict__ P,
                           int N, int K) {
  int idx = blockIdx.x * 256 + threadIdx.x;
  int total = (K >> 5) * (N >> 4) * 512;
  if (idx >= total) return;
  int i = idx & 15;
  int lane = (idx >> 4) & 31;
  int tile = idx >> 9;
  int ntiles = N >> 4;
  int kt = tile / ntiles, nt = tile - kt * ntiles;
  int n = nt * 16 + (lane & 15);
  int kb = (lane & 16) ? 8 : 0;
  int k = kt * 32 + kb + ((i < 8) ? i : (8 + i));
  P[idx] = (_Float16)W[(size_t)n * K + k];
}

// pack V (stored [b*64+k][256] f16) into per-(b,head) fragment tiles:
// P[(((b*8+h)*4) + kt*2 + nt)*512 + lane*16 + i]
__global__ void dat_pack_v(const _Float16* __restrict__ V,
                           _Float16* __restrict__ P) {
  int idx = blockIdx.x * 256 + threadIdx.x;   // 131072 total
  int i = idx & 15;
  int lane = (idx >> 4) & 31;
  int tile = idx >> 9;
  int nt = tile & 1, kt = (tile >> 1) & 1;
  int bh = tile >> 2;
  int h = bh & 7, b = bh >> 3;
  int n = h * 32 + nt * 16 + (lane & 15);
  int kb = (lane & 16) ? 8 : 0;
  int k = kt * 32 + kb + ((i < 8) ? i : (8 + i));
  P[idx] = V[((size_t)b * 64 + k) * 256 + n];
}

// ---------------------------------------------------------------------------
// K1: patch-embed conv (im2col GEMM, M=16 pix, N=256, K=192) + channel-LN
// feat stored pixel-major: feat[b][p][c]
// ---------------------------------------------------------------------------
__global__ __launch_bounds__(128) void dat_patch_embed(
    const float* __restrict__ x, const _Float16* __restrict__ wp,
    const float* __restrict__ pb, const float* __restrict__ lg,
    const float* __restrict__ lb, float* __restrict__ featF,
    _Float16* __restrict__ featH) {
  __shared__ float tile[16 * 257];
  __shared__ float mu[16], rs[16];
  const int b = blockIdx.y;
  const int p0 = blockIdx.x * 16;
  const int tid = threadIdx.x, wave = tid >> 5, lane = tid & 31;

  const int mrow = lane & 15;
  const int p = p0 + mrow;
  const int oh = p / 56, ow = p % 56;

  v8f zero = ZV8;
  v8f acc[4] = {zero, zero, zero, zero};

  for (int k0 = 0; k0 < 192; k0 += 32) {
    const int kb = k0 + ((lane & 16) ? 8 : 0);
    // each 8-run: fixed (c,kh), kw=0..7 -> 8 contiguous f32 (32B aligned)
    h8 run[2];
#pragma unroll
    for (int u = 0; u < 2; ++u) {
      int ks = kb + u * 16;
      int c = ks >> 6, rem = ks & 63, kh = rem >> 3;
      const float* xp =
          x + (((size_t)b * 3 + c) * 448 + oh * 8 + kh) * 448 + ow * 8;
      f4 a0 = *(const f4*)xp;
      f4 a1 = *(const f4*)(xp + 4);
      h8 r;
#pragma unroll
      for (int i = 0; i < 4; ++i) {
        r[i] = (_Float16)a0[i];
        r[i + 4] = (_Float16)a1[i];
      }
      run[u] = r;
    }
    v16h a = frag_from_runs(run[0], run[1]);
#pragma unroll
    for (int t = 0; t < 4; ++t) {
      v16h bf = load_frag_bp(wp, (size_t)(k0 >> 5) * 16 + wave * 4 + t, lane);
      acc[t] = wmma_f32(a, bf, acc[t]);
    }
  }
#pragma unroll
  for (int t = 0; t < 4; ++t)
#pragma unroll
    for (int r = 0; r < 8; ++r) {
      int m = r + ((lane & 16) ? 8 : 0);
      int n = wave * 64 + t * 16 + (lane & 15);
      tile[m * 257 + n] = acc[t][r] + pb[n];
    }
  __syncthreads();
  if (tid < 16) {
    float s = 0.f;
    for (int n = 0; n < 256; ++n) s += tile[tid * 257 + n];
    float m = s * (1.f / 256.f), s2 = 0.f;
    for (int n = 0; n < 256; ++n) {
      float d = tile[tid * 257 + n] - m;
      s2 += d * d;
    }
    mu[tid] = m;
    rs[tid] = rsqrtf(s2 * (1.f / 256.f) + 1e-5f);
  }
  __syncthreads();
  for (int i = tid * 4; i < 16 * 256; i += 128 * 4) {
    int m = i >> 8, n = i & 255;
    f4 o;
    h4 oh4;
#pragma unroll
    for (int j = 0; j < 4; ++j) {
      float v = (tile[m * 257 + n + j] - mu[m]) * rs[m] * lg[n + j] + lb[n + j];
      o[j] = v;
      oh4[j] = (_Float16)v;
    }
    size_t idx = ((size_t)b * 3136 + p0 + m) * 256 + n;
    *(f4*)(featF + idx) = o;
    *(h4*)(featH + idx) = oh4;
  }
}

// ---------------------------------------------------------------------------
// K2: generic WMMA GEMM: A (Mtot x K) f16 row-major, Bp packed frag tiles
// out[b*oB + p*oP + n*oN] (f32 and/or f16), m = b*Mb + p, N=256
// ---------------------------------------------------------------------------
__global__ __launch_bounds__(128) void dat_gemm256(
    const _Float16* __restrict__ A, const _Float16* __restrict__ Bp,
    const float* __restrict__ bias,
    float* outF, long long ofB, long long ofP, long long ofN,
    _Float16* outH, long long ohB, long long ohP, long long ohN,
    int Mb, int K) {
  const int tid = threadIdx.x, wave = tid >> 5, lane = tid & 31;
  const int m0 = blockIdx.x * 16;
  const int b = m0 / Mb, pbase = m0 - b * Mb;

  v8f zero = ZV8;
  v8f acc[4] = {zero, zero, zero, zero};
  const _Float16* Arow = A + (size_t)m0 * K;

  for (int k0 = 0; k0 < K; k0 += 32) {
    if (k0 + 32 < K) __builtin_prefetch(Arow + k0 + 32, 0, 1);
    v16h a = load_frag_a(Arow + k0, K, lane);
#pragma unroll
    for (int t = 0; t < 4; ++t) {
      v16h bf = load_frag_bp(Bp, (size_t)(k0 >> 5) * 16 + wave * 4 + t, lane);
      acc[t] = wmma_f32(a, bf, acc[t]);
    }
  }
#pragma unroll
  for (int t = 0; t < 4; ++t)
#pragma unroll
    for (int r = 0; r < 8; ++r) {
      int m = r + ((lane & 16) ? 8 : 0);
      int n = wave * 64 + t * 16 + (lane & 15);
      long long pp = pbase + m;
      float v = acc[t][r] + (bias ? bias[n] : 0.f);
      if (outF) outF[(long long)b * ofB + pp * ofP + (long long)n * ofN] = v;
      if (outH) outH[(long long)b * ohB + pp * ohP + (long long)n * ohN] = (_Float16)v;
    }
}

// ---------------------------------------------------------------------------
// K3: offset network (depthwise 28x28 s8 p14 + LN + GELU + pointwise + tanh)
// ---------------------------------------------------------------------------
__global__ __launch_bounds__(64) void dat_offset(
    const float* __restrict__ qF, const float* __restrict__ dww,
    const float* __restrict__ dwb, const float* __restrict__ lg,
    const float* __restrict__ lb, const float* __restrict__ pw,
    float* __restrict__ posW, float* __restrict__ posOut,
    float* __restrict__ refOut) {
  __shared__ float s[64], h2[64];
  const int blk = blockIdx.x;
  const int bg = blk >> 6, n = blk & 63;
  const int oh = n >> 3, ow = n & 7;
  const int b = bg >> 2, g = bg & 3;
  const int c = threadIdx.x;

  float a = dwb[c];
  for (int kh = 0; kh < 28; ++kh) {
    int ih = oh * 8 - 14 + kh;
    if (ih < 0 || ih >= 56) continue;
    for (int kw = 0; kw < 28; ++kw) {
      int iw = ow * 8 - 14 + kw;
      if (iw < 0 || iw >= 56) continue;
      a += dww[c * 784 + kh * 28 + kw] *
           qF[((size_t)b * 3136 + ih * 56 + iw) * 256 + g * 64 + c];
    }
  }
  s[c] = a;
  __syncthreads();
  float m = 0.f;
  for (int i = 0; i < 64; ++i) m += s[i];
  m *= (1.f / 64.f);
  float v = 0.f;
  for (int i = 0; i < 64; ++i) { float d = s[i] - m; v += d * d; }
  v *= (1.f / 64.f);
  float ln = (a - m) * rsqrtf(v + 1e-5f) * lg[c] + lb[c];
  h2[c] = 0.5f * ln * (1.f + erff(ln * 0.70710678118654752f));
  __syncthreads();
  if (c == 0) {
    float o0 = 0.f, o1 = 0.f;
    for (int i = 0; i < 64; ++i) { o0 += pw[i] * h2[i]; o1 += pw[64 + i] * h2[i]; }
    float offy = tanhf(o0) * (2.f / 7.f);
    float offx = tanhf(o1) * (2.f / 7.f);
    float ry = (0.5f + oh) * (2.f / 7.f) - 1.f;
    float rx = (0.5f + ow) * (2.f / 7.f) - 1.f;
    size_t base = (size_t)bg * 128 + n * 2;
    posW[base] = offy + ry;   posW[base + 1] = offx + rx;
    posOut[base] = offy + ry; posOut[base + 1] = offx + rx;
    refOut[base] = ry;        refOut[base + 1] = rx;
  }
}

// ---------------------------------------------------------------------------
// K4: bilinear sampling (align_corners=True, zero pad)
// block = (b, sample); 64 threads x 4 contiguous channels (float4 gathers)
// ---------------------------------------------------------------------------
__global__ __launch_bounds__(64) void dat_sample(
    const float* __restrict__ featF, const float* __restrict__ posW,
    _Float16* __restrict__ xsH) {
  const int blk = blockIdx.x;
  const int b = blk >> 6, n = blk & 63;
  const int c = threadIdx.x * 4;
  const int bg = b * 4 + (c >> 6);
  const float py = posW[(size_t)bg * 128 + n * 2];
  const float px = posW[(size_t)bg * 128 + n * 2 + 1];
  const float gx = (px + 1.f) * 27.5f;
  const float gy = (py + 1.f) * 27.5f;
  const int x0 = (int)floorf(gx), y0 = (int)floorf(gy);
  const float wx1 = gx - x0, wx0 = 1.f - wx1;
  const float wy1 = gy - y0, wy0 = 1.f - wy1;
  f4 acc = {0.f, 0.f, 0.f, 0.f};
#pragma unroll
  for (int q = 0; q < 4; ++q) {
    int ix = x0 + (q & 1), iy = y0 + (q >> 1);
    float w = ((q & 1) ? wx1 : wx0) * ((q >> 1) ? wy1 : wy0);
    if (ix >= 0 && ix <= 55 && iy >= 0 && iy <= 55) {
      f4 vv = *(const f4*)(featF + ((size_t)b * 3136 + iy * 56 + ix) * 256 + c);
#pragma unroll
      for (int j = 0; j < 4; ++j) acc[j] += w * vv[j];
    }
  }
  h4 o;
#pragma unroll
  for (int j = 0; j < 4; ++j) o[j] = (_Float16)acc[j];
  *(h4*)(xsH + ((size_t)b * 64 + n) * 256 + c) = o;
}

// ---------------------------------------------------------------------------
// K5: fused attention: S=QK^T (WMMA) + scale + rpe bias + softmax + P*V^T
// K-slice and packed-V tiles staged to LDS via async b128 copies (ASYNCcnt)
// ---------------------------------------------------------------------------
__global__ __launch_bounds__(128) void dat_attention(
    const _Float16* __restrict__ qH, const _Float16* __restrict__ kH,
    const _Float16* __restrict__ vP, const float* __restrict__ posW,
    const float* __restrict__ w1, const float* __restrict__ b1,
    const float* __restrict__ w2, _Float16* __restrict__ outH) {
  __shared__ __align__(16) _Float16 Ps[64 * 72];
  __shared__ __align__(16) _Float16 kT[64 * 32];   // 4KB, row-major ld=32
  __shared__ __align__(16) _Float16 vT[4 * 512];   // 4KB packed tiles
  __shared__ float w1s[64], b1s[32], w2s[32], poss[128];
  const int pt = blockIdx.x, h = blockIdx.y, b = blockIdx.z;
  const int gh = h & 1, bg = b * 4 + (h >> 1);
  const int p0 = pt * 64;
  const int tid = threadIdx.x, wave = tid >> 5, lane = tid & 31;

  // async stage K (64 samples x 32 head channels) and packed V (4 tiles)
  {
    const _Float16* kbase = kH + (size_t)b * 64 * 256 + h * 32;
#pragma unroll
    for (int u = 0; u < 2; ++u) {
      int chunk = tid + u * 128;            // 256 chunks of 16B
      int row = chunk >> 2, sub = chunk & 3;
      async_copy_b128((unsigned)(size_t)&kT[row * 32 + sub * 8],
                      kbase + (size_t)row * 256 + sub * 8);
    }
    const _Float16* vbase = vP + (size_t)(b * 8 + h) * 2048;
#pragma unroll
    for (int u = 0; u < 2; ++u) {
      int chunk = tid + u * 128;
      async_copy_b128((unsigned)(size_t)&vT[chunk * 8], vbase + chunk * 8);
    }
  }
  if (tid < 64) w1s[tid] = w1[tid];
  if (tid >= 64 && tid < 96) {
    b1s[tid - 64] = b1[tid - 64];
    w2s[tid - 64] = w2[gh * 32 + tid - 64];
  }
  poss[tid] = posW[(size_t)bg * 128 + tid];
  async_wait0();
  __syncthreads();

  v8f zero = ZV8;
  v8f acc[4] = {zero, zero, zero, zero};
  v16h qa = load_frag_a(qH + ((size_t)b * 3136 + p0 + wave * 16) * 256 + h * 32,
                        256, lane);
#pragma unroll
  for (int t = 0; t < 4; ++t) {
    v16h kf = load_frag_a(kT + t * 16 * 32, 32, lane);
    acc[t] = wmma_f32(qa, kf, acc[t]);
  }

  const float scale = 0.17677669529663687f;  // 32^-0.5
  const int lhalf = (lane & 16) ? 8 : 0;
  const int ln16 = lane & 15;
#pragma unroll
  for (int r = 0; r < 8; ++r) {
    const int m = wave * 16 + r + lhalf;
    const int p = p0 + m;
    const int qoh = p / 56, qow = p % 56;
    const float qy = (float)qoh * (56.f / 55.f) * (2.f / 55.f) - 1.f;
    const float qx = (float)qow * (56.f / 55.f) * (2.f / 55.f) - 1.f;
    float e[4];
    float mx = -1e30f;
#pragma unroll
    for (int t = 0; t < 4; ++t) {
      const int n = t * 16 + ln16;
      float dy = (qy - poss[n * 2]) * 4.f;
      float dx = (qx - poss[n * 2 + 1]) * 4.f;
      float ly = copysignf(log2f(fabsf(dy) + 1.f) * (1.f / 3.f), dy);
      float lx = copysignf(log2f(fabsf(dx) + 1.f) * (1.f / 3.f), dx);
      float bv = 0.f;
      for (int j = 0; j < 32; ++j) {
        float hh = fmaxf(w1s[2 * j] * ly + w1s[2 * j + 1] * lx + b1s[j], 0.f);
        bv += w2s[j] * hh;
      }
      e[t] = acc[t][r] * scale + bv;
      mx = fmaxf(mx, e[t]);
    }
#pragma unroll
    for (int o = 1; o < 16; o <<= 1) mx = fmaxf(mx, __shfl_xor(mx, o, 32));
    float sum = 0.f;
#pragma unroll
    for (int t = 0; t < 4; ++t) { e[t] = expf(e[t] - mx); sum += e[t]; }
#pragma unroll
    for (int o = 1; o < 16; o <<= 1) sum += __shfl_xor(sum, o, 32);
    const float inv = 1.f / sum;
#pragma unroll
    for (int t = 0; t < 4; ++t)
      Ps[m * 72 + t * 16 + ln16] = (_Float16)(e[t] * inv);
  }
  __syncthreads();

  // out = P (64x64) * V^T (64x32), V tiles from LDS
  v8f oacc[2] = {zero, zero};
#pragma unroll
  for (int kk = 0; kk < 2; ++kk) {
    v16h pa = load_frag_a(&Ps[(wave * 16) * 72 + kk * 32], 72, lane);
#pragma unroll
    for (int nt = 0; nt < 2; ++nt) {
      v16h vf = *(const v16h*)(vT + (kk * 2 + nt) * 512 + lane * 16);
      oacc[nt] = wmma_f32(pa, vf, oacc[nt]);
    }
  }
#pragma unroll
  for (int nt = 0; nt < 2; ++nt)
#pragma unroll
    for (int r = 0; r < 8; ++r) {
      const int m = wave * 16 + r + lhalf;
      const int cc = h * 32 + nt * 16 + ln16;
      outH[((size_t)b * 3136 + p0 + m) * 256 + cc] = (_Float16)oacc[nt][r];
    }
}

// ---------------------------------------------------------------------------
extern "C" void kernel_launch(void* const* d_in, const int* in_sizes, int n_in,
                              void* d_out, int out_size, void* d_ws, size_t ws_size,
                              hipStream_t stream) {
  (void)in_sizes; (void)n_in; (void)out_size; (void)ws_size;
  const float* x        = (const float*)d_in[0];
  const float* patch_w  = (const float*)d_in[1];
  const float* patch_b  = (const float*)d_in[2];
  const float* ln1_g    = (const float*)d_in[3];
  const float* ln1_b    = (const float*)d_in[4];
  const float* wq = (const float*)d_in[5];  const float* bq = (const float*)d_in[6];
  const float* wk = (const float*)d_in[7];  const float* bk = (const float*)d_in[8];
  const float* wv = (const float*)d_in[9];  const float* bv = (const float*)d_in[10];
  const float* wo = (const float*)d_in[11]; const float* bo = (const float*)d_in[12];
  const float* off_dw_w = (const float*)d_in[13];
  const float* off_dw_b = (const float*)d_in[14];
  const float* off_ln_g = (const float*)d_in[15];
  const float* off_ln_b = (const float*)d_in[16];
  const float* off_pw_w = (const float*)d_in[17];
  const float* rpe_w1   = (const float*)d_in[18];
  const float* rpe_b1   = (const float*)d_in[19];
  const float* rpe_w2   = (const float*)d_in[20];

  char* ws = (char*)d_ws;
  size_t off = 0;
  auto alloc = [&](size_t bytes) -> void* {
    void* p = ws + off;
    off += (bytes + 255) & ~(size_t)255;
    return p;
  };
  const size_t NPC = (size_t)8 * 3136 * 256;        // B*HW*C
  float*    featF = (float*)alloc(NPC * 4);
  _Float16* featH = (_Float16*)alloc(NPC * 2);
  float*    qF    = (float*)alloc(NPC * 4);
  _Float16* qH    = (_Float16*)alloc(NPC * 2);
  _Float16* outH  = (_Float16*)alloc(NPC * 2);
  _Float16* xsH   = (_Float16*)alloc((size_t)8 * 64 * 256 * 2);
  _Float16* kHb   = (_Float16*)alloc((size_t)8 * 64 * 256 * 2);
  _Float16* vHb   = (_Float16*)alloc((size_t)8 * 64 * 256 * 2);
  _Float16* vPck  = (_Float16*)alloc((size_t)131072 * 2);
  float*    posW  = (float*)alloc((size_t)32 * 64 * 2 * 4);
  _Float16* wqP   = (_Float16*)alloc(65536 * 2);
  _Float16* wkP   = (_Float16*)alloc(65536 * 2);
  _Float16* wvP   = (_Float16*)alloc(65536 * 2);
  _Float16* woP   = (_Float16*)alloc(65536 * 2);
  _Float16* wpP   = (_Float16*)alloc(49152 * 2);

  float* yOut   = (float*)d_out;                    // (B,256,56,56)
  float* posOut = yOut + (size_t)8 * 256 * 3136;    // (B,4,8,8,2)
  float* refOut = posOut + 4096;                    // (B,4,8,8,2)

  // K0: pack weights into WMMA fragment tiles
  dat_pack_b<<<512, 256, 0, stream>>>(wq, wqP, 256, 256);
  dat_pack_b<<<512, 256, 0, stream>>>(wk, wkP, 256, 256);
  dat_pack_b<<<512, 256, 0, stream>>>(wv, wvP, 256, 256);
  dat_pack_b<<<512, 256, 0, stream>>>(wo, woP, 256, 256);
  dat_pack_b<<<192, 256, 0, stream>>>(patch_w, wpP, 256, 192);

  // K1: patch embed + LN
  dat_patch_embed<<<dim3(196, 8), 128, 0, stream>>>(x, wpP, patch_b, ln1_g, ln1_b,
                                                    featF, featH);
  // K2a: q projection (f32 for offset net, f16 for attention)
  dat_gemm256<<<25088 / 16, 128, 0, stream>>>(featH, wqP, bq,
      qF, (long long)3136 * 256, 256, 1,
      qH, (long long)3136 * 256, 256, 1, 3136, 256);
  // K3: offset network -> pos / ref
  dat_offset<<<2048, 64, 0, stream>>>(qF, off_dw_w, off_dw_b, off_ln_g, off_ln_b,
                                      off_pw_w, posW, posOut, refOut);
  // K4: deformable bilinear sampling
  dat_sample<<<512, 64, 0, stream>>>(featF, posW, xsH);
  // K2b/c: k and v projections
  dat_gemm256<<<512 / 16, 128, 0, stream>>>(xsH, wkP, bk,
      nullptr, 0, 0, 0, kHb, (long long)64 * 256, 256, 1, 64, 256);
  dat_gemm256<<<512 / 16, 128, 0, stream>>>(xsH, wvP, bv,
      nullptr, 0, 0, 0, vHb, (long long)64 * 256, 256, 1, 64, 256);
  // pack V into per-(b,head) fragment tiles
  dat_pack_v<<<512, 256, 0, stream>>>(vHb, vPck);
  // K5: fused attention (async-LDS staging of K/V)
  dat_attention<<<dim3(49, 8, 8), 128, 0, stream>>>(qH, kHb, vPck, posW,
                                                    rpe_w1, rpe_b1, rpe_w2, outH);
  // K2d: output projection, scattered to (B, C, HW)
  dat_gemm256<<<25088 / 16, 128, 0, stream>>>(outH, woP, bo,
      yOut, (long long)256 * 3136, 1, 3136,
      nullptr, 0, 0, 0, 3136, 256);
}
